// LinearAttentionModel_7834020348444
// MI455X (gfx1250) — matrix operand order repaired and checked by reading, not compiled, not run on hardware
//
#include <hip/hip_runtime.h>

// ---------------------------------------------------------------------------
// Model dims (fixed by the reference)
// ---------------------------------------------------------------------------
#define DIM    1024
#define VOCAB  32000
#define SEQ    2048
#define BATCH  2
#define ROWS   (BATCH * SEQ)   // 4096
#define CTX    4
#define EPS_A  1e-6f
#define LN_EPS 1e-5f

typedef __bf16 bf16_t;
typedef __attribute__((ext_vector_type(16))) __bf16 v16bf;
typedef __attribute__((ext_vector_type(8)))  __bf16 v8bf;
typedef __attribute__((ext_vector_type(8)))  float  v8f;

// ---------------------------------------------------------------------------
// WMMA fragment helpers (wave32, V_WMMA_F32_16X16X32_BF16)
//
// A operand (16x32, 16-bit), contiguous k order used here (not the ISA table order):
//   lanes 0-15 : M = lane,     K = [0..15]
//   lanes 16-31: M = lane-16,  K = [16..31]
// -> per lane: one contiguous 16-element run (two 16B loads).
//
// B operand (32x16, 16-bit), K-contiguous per lane:
//   lanes 0-15 : N = lane,     K = [0..15]
//   lanes 16-31: N = lane-16,  K = [16..31]
// -> per lane: one contiguous 16-element chunk (one 32B load).
// Every B matrix is stored transposed (row = N, K contiguous).
// ---------------------------------------------------------------------------
__device__ __forceinline__ v16bf load_a_frag(const bf16_t* base, int ld, int row0, int k0) {
  int lane = threadIdx.x & 31;
  int r    = row0 + (lane & 15);
  int koff = (lane >> 4) * 16;
  const bf16_t* p = base + (size_t)r * ld + k0 + koff;
  v8bf lo = *(const v8bf*)p;
  v8bf hi = *(const v8bf*)(p + 8);
  return __builtin_shufflevector(lo, hi, 0,1,2,3,4,5,6,7,8,9,10,11,12,13,14,15);
}

__device__ __forceinline__ v16bf load_b_frag(const bf16_t* base, int ld, int col0, int k0) {
  int lane = threadIdx.x & 31;
  int c    = col0 + (lane & 15);
  int k    = k0 + (lane >> 4) * 16;
  return *(const v16bf*)(base + (size_t)c * ld + k);
}

__device__ __forceinline__ v8f wmma_bf16(v16bf a, v16bf b, v8f c) {
  // (neg_a, A, neg_b, B, c_mod, C, reuse_a, reuse_b)
  return __builtin_amdgcn_wmma_f32_16x16x32_bf16(false, a, false, b, (short)0, c,
                                                 false, false);
}

__device__ __forceinline__ v8f v8f_zero() {
  v8f z = {0.f, 0.f, 0.f, 0.f, 0.f, 0.f, 0.f, 0.f};
  return z;
}

// ---------------------------------------------------------------------------
// K0: transpose + convert f32 -> bf16.  in: [R][C] f32, out: [C][R] bf16
// ---------------------------------------------------------------------------
__global__ void __launch_bounds__(256)
transpose_bf16(const float* __restrict__ in, bf16_t* __restrict__ out, int R, int C) {
  __shared__ float tile[32][33];
  int c0 = blockIdx.x * 32, r0 = blockIdx.y * 32;
  int tx = threadIdx.x & 31, ty = threadIdx.x >> 5;  // 32 x 8
  for (int i = ty; i < 32; i += 8) {
    int r = r0 + i, c = c0 + tx;
    tile[i][tx] = (r < R && c < C) ? in[(size_t)r * C + c] : 0.f;
  }
  __syncthreads();
  for (int i = ty; i < 32; i += 8) {
    int c = c0 + i, r = r0 + tx;
    if (c < C && r < R) out[(size_t)c * R + r] = (bf16_t)tile[tx][i];
  }
}

// ---------------------------------------------------------------------------
// K1: embedding gather + local-context sum (CTX taps) -> bf16 activations
// ---------------------------------------------------------------------------
__global__ void __launch_bounds__(256)
embed_ctx(const int* __restrict__ x, const float* __restrict__ E,
          bf16_t* __restrict__ embB) {
  size_t idx = (size_t)blockIdx.x * blockDim.x + threadIdx.x;  // over B*S*D
  int d = (int)(idx % DIM);
  int s = (int)((idx / DIM) % SEQ);
  int b = (int)(idx / ((size_t)DIM * SEQ));
  float acc = 0.f;
#pragma unroll
  for (int o = 0; o < CTX; ++o) {
    int sp = s - o;
    if (sp >= 0) {
      int tok = x[b * SEQ + sp];
      acc += E[(size_t)tok * DIM + d];
    }
  }
  embB[idx] = (bf16_t)acc;
}

// ---------------------------------------------------------------------------
// K2: generic 64x256-tile bf16 WMMA GEMM.  C = A[M,K] * B (B stored as Bt[N][K]).
// 256 threads = 8 waves; wave w owns N columns [n0+32w, n0+32w+32): amortizes
// the 4 shared A fragments over 8 WMMAs (2 B frags) -> ~0.5KB operand/WMMA.
// MODE 0: +bias, elu(x)+1, bf16 out    (q, k)
// MODE 1: +bias, bf16 out              (v)
// MODE 2: +bias, f32 out               (out-proj, logits)
// ---------------------------------------------------------------------------
template <int MODE>
__global__ void __launch_bounds__(256)
gemm64x256(const bf16_t* __restrict__ A, const bf16_t* __restrict__ Bt,
           const float* __restrict__ bias, void* __restrict__ out,
           int N, int K) {
  int m0   = blockIdx.x * 64;
  int n0   = blockIdx.y * 256;
  int lane = threadIdx.x & 31;
  int wave = (int)(threadIdx.x >> 5);
  int nw   = n0 + wave * 32;

  v8f acc[8];
#pragma unroll
  for (int i = 0; i < 8; ++i) acc[i] = v8f_zero();

  for (int k0 = 0; k0 < K; k0 += 32) {
    // speculative prefetch two k-steps ahead (global_prefetch_b8)
    if (k0 + 64 < K) {
      __builtin_prefetch(Bt + (size_t)(nw + (lane & 15)) * K + k0 + 64);
      __builtin_prefetch(A + (size_t)(m0 + (lane & 15)) * K + k0 + 64);
      __builtin_prefetch(A + (size_t)(m0 + 32 + (lane & 15)) * K + k0 + 64);
    }
    v16bf bf0 = load_b_frag(Bt, K, nw,      k0);
    v16bf bf1 = load_b_frag(Bt, K, nw + 16, k0);
#pragma unroll
    for (int mi = 0; mi < 4; ++mi) {
      v16bf af = load_a_frag(A, K, m0 + mi * 16, k0);
      acc[mi * 2 + 0] = wmma_bf16(af, bf0, acc[mi * 2 + 0]);
      acc[mi * 2 + 1] = wmma_bf16(af, bf1, acc[mi * 2 + 1]);
    }
  }

  // C/D layout: VGPR j -> M=j (lanes 0-15) / M=j+8 (lanes 16-31), N = lane&15
  int rhi = (lane >> 4) * 8;
#pragma unroll
  for (int t2 = 0; t2 < 2; ++t2) {
    int   col = nw + t2 * 16 + (lane & 15);
    float bv  = bias[col];
#pragma unroll
    for (int mi = 0; mi < 4; ++mi) {
#pragma unroll
      for (int j = 0; j < 8; ++j) {
        int   row = m0 + mi * 16 + j + rhi;
        float t   = acc[mi * 2 + t2][j] + bv;
        if (MODE == 0) t = (t > 0.f) ? (t + 1.f) : __expf(t);  // elu(x)+1
        if (MODE == 2)
          ((float*)out)[(size_t)row * N + col] = t;
        else
          ((bf16_t*)out)[(size_t)row * N + col] = (bf16_t)t;
      }
    }
  }
}

// ---------------------------------------------------------------------------
// K3: causal linear attention (quadratic tiled form, all-WMMA).
// grid = (SEQ/64 q-tiles, DIM/256 v-slices, BATCH); 256 threads = 8 waves.
// Per k-tile:  S = q_tile . k_tile^T  (WMMA, K=DIM)  -> masked bf16 in LDS
//              num += S @ V_slice     (WMMA from LDS, K=64)
//              den += rowsum(S)       (from the same rounded S tile)
// Epilogue: attn = num / (den + eps) -> bf16.
// ---------------------------------------------------------------------------
#define VT_LD 80  // padded LDS stride for transposed V tile (keeps 32B alignment)

__global__ void __launch_bounds__(256)
attn_kernel(const bf16_t* __restrict__ q, const bf16_t* __restrict__ k,
            const bf16_t* __restrict__ v, bf16_t* __restrict__ attn) {
  int qt  = blockIdx.x;        // q tile (64 rows)
  int e0  = blockIdx.y * 256;  // v-column slice
  int b   = blockIdx.z;
  int tid = threadIdx.x, lane = tid & 31, wave = tid >> 5;
  int m0  = qt * 64;

  __shared__ __align__(32) bf16_t s_lds[64 * 64];       //  8 KB
  __shared__ __align__(32) bf16_t vt_lds[256 * VT_LD];  // 40 KB
  __shared__ float den_lds[64];

  const bf16_t* qb = q + (size_t)b * SEQ * DIM;
  const bf16_t* kb = k + (size_t)b * SEQ * DIM;
  const bf16_t* vb = v + (size_t)b * SEQ * DIM;

  if (tid < 64) den_lds[tid] = 0.f;

  v8f acc[8];
#pragma unroll
  for (int i = 0; i < 8; ++i) acc[i] = v8f_zero();

  // score sub-tile ownership: wave -> (mi = wave/2, ni = 2*(wave&1) + {0,1})
  int smi = wave >> 1;
  int sni = (wave & 1) * 2;

  for (int kt = 0; kt <= qt; ++kt) {
    int kr0 = kt * 64;

    // ---- scores: one shared A-frag, two B-frags per k-step ----------------
    {
      v8f s0 = v8f_zero(), s1 = v8f_zero();
      for (int k0 = 0; k0 < DIM; k0 += 32) {
        v16bf af  = load_a_frag(qb, DIM, m0 + smi * 16, k0);
        v16bf bf0 = load_b_frag(kb, DIM, kr0 + (sni + 0) * 16, k0);
        v16bf bf1 = load_b_frag(kb, DIM, kr0 + (sni + 1) * 16, k0);
        s0 = wmma_bf16(af, bf0, s0);
        s1 = wmma_bf16(af, bf1, s1);
      }
      int rhi = (lane >> 4) * 8;
#pragma unroll
      for (int t = 0; t < 2; ++t) {
        v8f& s    = (t == 0) ? s0 : s1;
        int  coll = (sni + t) * 16 + (lane & 15);
#pragma unroll
        for (int j = 0; j < 8; ++j) {
          int   rowl = smi * 16 + j + rhi;
          float sv   = s[j];
          if (kr0 + coll > m0 + rowl) sv = 0.f;  // causal mask (diagonal tile)
          s_lds[rowl * 64 + coll] = (bf16_t)sv;
        }
      }
    }

    // ---- stage V tile transposed: vt_lds[e_local][n] ----------------------
    {
      const bf16_t* vp = vb + (size_t)kr0 * DIM + e0 + tid;
      for (int n = 0; n < 64; ++n)
        vt_lds[tid * VT_LD + n] = vp[(size_t)n * DIM];
    }
    __syncthreads();

    // ---- denominator from the same rounded scores -------------------------
    if (tid < 64) {
      float rs = 0.f;
      for (int n = 0; n < 64; ++n) rs += (float)s_lds[tid * 64 + n];
      den_lds[tid] += rs;
    }

    // ---- num += S @ V_slice : wave owns 32 cols, 4 M x 2 N sub-tiles ------
#pragma unroll
    for (int kk = 0; kk < 64; kk += 32) {
#pragma unroll
      for (int t2 = 0; t2 < 2; ++t2) {
        v16bf bf = load_b_frag(vt_lds, VT_LD, wave * 32 + t2 * 16, kk);
#pragma unroll
        for (int mi = 0; mi < 4; ++mi) {
          v16bf af = load_a_frag(s_lds, 64, mi * 16, kk);
          acc[mi * 2 + t2] = wmma_bf16(af, bf, acc[mi * 2 + t2]);
        }
      }
    }
    __syncthreads();  // protects s_lds / vt_lds reuse; also den_lds visibility
  }

  // ---- epilogue: divide by denominator, write bf16 ------------------------
  bf16_t* ab  = attn + (size_t)b * SEQ * DIM;
  int     rhi = (lane >> 4) * 8;
#pragma unroll
  for (int mi = 0; mi < 4; ++mi) {
#pragma unroll
    for (int t2 = 0; t2 < 2; ++t2) {
      int colg = e0 + wave * 32 + t2 * 16 + (lane & 15);
#pragma unroll
      for (int j = 0; j < 8; ++j) {
        int   rowl = mi * 16 + j + rhi;
        float d    = den_lds[rowl] + EPS_A;
        ab[(size_t)(m0 + rowl) * DIM + colg] = (bf16_t)(acc[mi * 2 + t2][j] / d);
      }
    }
  }
}

// ---------------------------------------------------------------------------
// K5: LayerNorm over DIM, one row per workgroup -> bf16 (feeds logits GEMM)
// ---------------------------------------------------------------------------
__global__ void __launch_bounds__(256)
ln_kernel(const float* __restrict__ in, const float* __restrict__ gamma,
          const float* __restrict__ beta, bf16_t* __restrict__ out) {
  int row = blockIdx.x;
  int tid = threadIdx.x;
  __shared__ float red[256];
  const float* r = in + (size_t)row * DIM;

  float vals[4];
  float s = 0.f;
#pragma unroll
  for (int i = 0; i < 4; ++i) { vals[i] = r[tid + i * 256]; s += vals[i]; }
  red[tid] = s;
  __syncthreads();
  for (int off = 128; off > 0; off >>= 1) {
    if (tid < off) red[tid] += red[tid + off];
    __syncthreads();
  }
  float mu = red[0] / DIM;
  __syncthreads();

  float vs = 0.f;
#pragma unroll
  for (int i = 0; i < 4; ++i) { float d = vals[i] - mu; vs += d * d; }
  red[tid] = vs;
  __syncthreads();
  for (int off = 128; off > 0; off >>= 1) {
    if (tid < off) red[tid] += red[tid + off];
    __syncthreads();
  }
  float inv = rsqrtf(red[0] / DIM + LN_EPS);

#pragma unroll
  for (int i = 0; i < 4; ++i) {
    int c = tid + i * 256;
    out[(size_t)row * DIM + c] = (bf16_t)((vals[i] - mu) * inv * gamma[c] + beta[c]);
  }
}

// ---------------------------------------------------------------------------
// kernel_launch
// ---------------------------------------------------------------------------
extern "C" void kernel_launch(void* const* d_in, const int* in_sizes, int n_in,
                              void* d_out, int out_size, void* d_ws, size_t ws_size,
                              hipStream_t stream) {
  const int*   x    = (const int*)  d_in[0];
  const float* E    = (const float*)d_in[1];
  const float* Wq   = (const float*)d_in[2];
  const float* bq   = (const float*)d_in[3];
  const float* Wk   = (const float*)d_in[4];
  const float* bk   = (const float*)d_in[5];
  const float* Wv   = (const float*)d_in[6];
  const float* bv   = (const float*)d_in[7];
  const float* Wo   = (const float*)d_in[8];
  const float* bo   = (const float*)d_in[9];
  const float* lng  = (const float*)d_in[10];
  const float* lnb  = (const float*)d_in[11];
  const float* Wout = (const float*)d_in[12];
  const float* bout = (const float*)d_in[13];
  (void)in_sizes; (void)n_in; (void)out_size; (void)ws_size;

  char*  ws  = (char*)d_ws;
  size_t off = 0;
  auto alloc = [&](size_t bytes) -> void* {
    void* p = ws + off;
    off += (bytes + 255) & ~(size_t)255;
    return p;
  };

  bf16_t* WqT   = (bf16_t*)alloc((size_t)DIM * DIM * 2);
  bf16_t* WkT   = (bf16_t*)alloc((size_t)DIM * DIM * 2);
  bf16_t* WvT   = (bf16_t*)alloc((size_t)DIM * DIM * 2);
  bf16_t* WoT   = (bf16_t*)alloc((size_t)DIM * DIM * 2);
  bf16_t* WoutT = (bf16_t*)alloc((size_t)VOCAB * DIM * 2);
  bf16_t* embB  = (bf16_t*)alloc((size_t)ROWS * DIM * 2);
  bf16_t* qB    = (bf16_t*)alloc((size_t)ROWS * DIM * 2);
  bf16_t* kB    = (bf16_t*)alloc((size_t)ROWS * DIM * 2);
  bf16_t* vB    = (bf16_t*)alloc((size_t)ROWS * DIM * 2);
  bf16_t* attnB = (bf16_t*)alloc((size_t)ROWS * DIM * 2);
  float*  outF  = (float*) alloc((size_t)ROWS * DIM * 4);
  bf16_t* lnB   = (bf16_t*)alloc((size_t)ROWS * DIM * 2);

  dim3 blk(256);

  // weights -> bf16, transposed (K-contiguous B operands)
  dim3 gT(DIM / 32, DIM / 32);
  transpose_bf16<<<gT, blk, 0, stream>>>(Wq,   WqT,   DIM, DIM);
  transpose_bf16<<<gT, blk, 0, stream>>>(Wk,   WkT,   DIM, DIM);
  transpose_bf16<<<gT, blk, 0, stream>>>(Wv,   WvT,   DIM, DIM);
  transpose_bf16<<<gT, blk, 0, stream>>>(Wo,   WoT,   DIM, DIM);
  transpose_bf16<<<dim3(VOCAB / 32, DIM / 32), blk, 0, stream>>>(Wout, WoutT, DIM, VOCAB);

  // embedding gather + context sum
  embed_ctx<<<(ROWS * DIM) / 256, blk, 0, stream>>>(x, E, embB);

  // q / k / v projections
  dim3 gP(ROWS / 64, DIM / 256);
  gemm64x256<0><<<gP, blk, 0, stream>>>(embB, WqT, bq, qB, DIM, DIM);
  gemm64x256<0><<<gP, blk, 0, stream>>>(embB, WkT, bk, kB, DIM, DIM);
  gemm64x256<1><<<gP, blk, 0, stream>>>(embB, WvT, bv, vB, DIM, DIM);

  // causal linear attention
  attn_kernel<<<dim3(SEQ / 64, DIM / 256, BATCH), blk, 0, stream>>>(qB, kB, vB, attnB);

  // out-proj (f32), LayerNorm (bf16)
  gemm64x256<2><<<gP, blk, 0, stream>>>(attnB, WoT, bo, outF, DIM, DIM);
  ln_kernel<<<ROWS, blk, 0, stream>>>(outF, lng, lnb, lnB);

  // logits: [4096,1024] x [1024,32000] -> f32 d_out
  gemm64x256<2><<<dim3(ROWS / 64, VOCAB / 256), blk, 0, stream>>>(
      lnB, WoutT, bout, (float*)d_out, VOCAB, DIM);
}
